// Selector_1546188226591
// MI455X (gfx1250) — matrix operand order repaired and checked by reading, not compiled, not run on hardware
//
#include <hip/hip_runtime.h>
#include <math.h>

#define BSZ    256
#define LSEQ   64
#define HDIM   2048
#define NITEMS 50000
#define KTOP   20

typedef __attribute__((ext_vector_type(16))) __bf16    v16bf;
typedef __attribute__((ext_vector_type(8)))  float     v8f;
typedef __attribute__((ext_vector_type(8)))  unsigned  v8u;

// pack bf16(x) (low 16) | bf16(y) (high 16) with one v_perm_b32
__device__ __forceinline__ unsigned pack2(float x, float y) {
  return __builtin_amdgcn_perm(__builtin_bit_cast(unsigned, y),
                               __builtin_bit_cast(unsigned, x), 0x07060302u);
}

// 16 consecutive f32 (four float4) -> v16bf, element i = bf16 of float i
__device__ __forceinline__ v16bf cvt16(float4 a, float4 b, float4 c, float4 d) {
  v8u r;
  r[0] = pack2(a.x, a.y); r[1] = pack2(a.z, a.w);
  r[2] = pack2(b.x, b.y); r[3] = pack2(b.z, b.w);
  r[4] = pack2(c.x, c.y); r[5] = pack2(c.z, c.w);
  r[6] = pack2(d.x, d.y); r[7] = pack2(d.z, d.w);
  return __builtin_bit_cast(v16bf, r);
}

// A fragment 16x32 (MxK), 16-bit layout (ISA 7.12.2):
//   lanes 0-15 : elems 0-7 = K 0..7,   elems 8-15 = K 16..23
//   lanes 16-31: elems 0-7 = K 8..15,  elems 8-15 = K 24..31
__device__ __forceinline__ v16bf load_a_frag(const float* rowp, int lane) {
  int koff = (lane & 16) ? 8 : 0;
  const float4* p0 = (const float4*)(rowp + koff);
  const float4* p1 = (const float4*)(rowp + koff + 16);
  return cvt16(p0[0], p0[1], p1[0], p1[1]);
}

// low 32 bits of a generic LDS pointer == wave-relative LDS byte address
__device__ __forceinline__ unsigned lds_off(const void* p) {
  return (unsigned)(unsigned long long)p;
}

// single async DMA copy, 16B, shared literal offset applies to LDS + global addr
#define ASYNC16(l, g, OFF)                                                    \
  asm volatile("global_load_async_to_lds_b128 %0, %1, off offset:" #OFF       \
               :: "v"(l), "v"(g) : "memory");

// C[row, col] = sum_k A[row, k] * B[col, k]   ("NT" GEMM, both row-major along K)
// Block tile 128x128, 8 waves; B tile staged in LDS via async DMA, double-buffered.
template<bool CHECK_N>
__device__ __forceinline__ void gemm_nt_128x128(
    const float* __restrict__ A, const float* __restrict__ Bm,
    float* __restrict__ C, int N, int K, int mBlk, int nBlk) {
  constexpr int CSTRIDE = 36;            // 32 + 4 pad floats per staged column
  __shared__ float bbuf[2][128 * CSTRIDE];

  const int tid  = threadIdx.x;
  const int lane = tid & 31;
  const int wave = tid >> 5;
  const int wm = wave >> 1, wn = wave & 1;
  const int l15 = lane & 15;
  const int rowBase = mBlk * 128 + wm * 32;
  const int colBase = nBlk * 128;

  const float* aRow0 = A + (size_t)(rowBase + l15) * HDIM;  // K == HDIM here
  const float* aRow1 = aRow0 + (size_t)16 * HDIM;

  // ---- async staging roles: thread t copies 4 x 16B for column (t>>1) ----
  const int ccol = tid >> 1;             // 0..127  (staged column)
  const int foff = (tid & 1) * 16;       // float offset within the 32-K slab
  int gcol = colBase + ccol;
  if (CHECK_N && gcol >= N) gcol = 0;    // clamp; garbage cols never stored
  const float* gsrc = Bm + (size_t)gcol * K + foff;
  unsigned ldst[2];
  ldst[0] = lds_off(&bbuf[0][ccol * CSTRIDE + foff]);
  ldst[1] = lds_off(&bbuf[1][ccol * CSTRIDE + foff]);

  #define PREFETCH_B(p, k0)                                                   \
    {                                                                         \
      const float* g = gsrc + (k0);                                           \
      unsigned l = ldst[p];                                                   \
      ASYNC16(l, g, 0)  ASYNC16(l, g, 16)                                     \
      ASYNC16(l, g, 32) ASYNC16(l, g, 48)                                     \
    }

  const int bko = (lane & 16) ? 8 : 0;   // K-octet select for B fragment reads (table order)

  v8f acc[2][4] = {};
  PREFETCH_B(0, 0)
  int p = 0;
  for (int k0 = 0; k0 < K; k0 += 32) {
    if (k0 + 32 < K) {
      PREFETCH_B(p ^ 1, k0 + 32)
      asm volatile("s_wait_asynccnt 0x4" ::: "memory");  // my copies into buf[p] done
    } else {
      asm volatile("s_wait_asynccnt 0x0" ::: "memory");
    }
    __syncthreads();                                     // everyone's copies done

    __builtin_prefetch(aRow0 + k0 + 256, 0, 3);
    __builtin_prefetch(aRow1 + k0 + 256, 0, 3);
    v16bf a0 = load_a_frag(aRow0 + k0, lane);
    v16bf a1 = load_a_frag(aRow1 + k0, lane);

    // build all four B fragments first, then issue the 8 WMMAs back-to-back:
    // removes the WMMA->v_perm WAR hazard nops inside the j loop.
    const float* bb = bbuf[p];
    v16bf bf[4];
#pragma unroll
    for (int j = 0; j < 4; ++j) {
      const float4* q =
          (const float4*)(bb + (wn * 64 + j * 16 + l15) * CSTRIDE + bko);
      bf[j] = cvt16(q[0], q[1], q[4], q[5]);
    }
#pragma unroll
    for (int j = 0; j < 4; ++j) {
      acc[0][j] = __builtin_amdgcn_wmma_f32_16x16x32_bf16(
          false, a0, false, bf[j], (short)0, acc[0][j], false, false);
      acc[1][j] = __builtin_amdgcn_wmma_f32_16x16x32_bf16(
          false, a1, false, bf[j], (short)0, acc[1][j], false, false);
    }
    __syncthreads();                    // reads done before buf[p] is re-filled
    p ^= 1;
  }
  #undef PREFETCH_B

  // C/D layout: col = lane&15; VGPR r -> row r (lanes 0-15) or r+8 (lanes 16-31)
  const int rOff = (lane >> 4) * 8;
#pragma unroll
  for (int i = 0; i < 2; ++i) {
#pragma unroll
    for (int j = 0; j < 4; ++j) {
      int c = colBase + wn * 64 + j * 16 + l15;
      if (CHECK_N && c >= N) continue;
#pragma unroll
      for (int r = 0; r < 8; ++r) {
        int row = rowBase + i * 16 + rOff + r;
        C[(size_t)row * N + c] = acc[i][j][r];
      }
    }
  }
}

// A_gen/B_gen = gen @ proj{A,B}^T : [16384,2048] x [2048,2048]
// grid: x = N tiles (fastest) so the L2-resident 16MB weight matrix is reused.
__global__ __launch_bounds__(256)
void proj_kernel(const float* __restrict__ gen,
                 const float* __restrict__ pA, const float* __restrict__ pB,
                 float* __restrict__ Ag, float* __restrict__ Bg) {
  const float* W = blockIdx.z ? pB : pA;
  float* C = blockIdx.z ? Bg : Ag;
  gemm_nt_128x128<false>(gen, W, C, HDIM, HDIM, blockIdx.y, blockIdx.x);
}

// attention pooling per (batch, domain); also emits pooled-vector L2 norm
__global__ __launch_bounds__(256)
void pool_kernel(const float* __restrict__ Agen, const float* __restrict__ Bgen,
                 const float* __restrict__ wA, const float* __restrict__ wB,
                 float* __restrict__ pooledA, float* __restrict__ pooledB,
                 float* __restrict__ pnA, float* __restrict__ pnB) {
  const int b = blockIdx.x;
  const float* x = (blockIdx.y == 0 ? Agen : Bgen) + (size_t)b * LSEQ * HDIM;
  const float* w = blockIdx.y == 0 ? wA : wB;
  float* pooled  = (blockIdx.y == 0 ? pooledA : pooledB) + (size_t)b * HDIM;
  float* pn      = blockIdx.y == 0 ? pnA : pnB;

  __shared__ float attn[LSEQ];
  __shared__ float nrm;
  const int tid = threadIdx.x, lane = tid & 31, wv = tid >> 5;

  for (int l = wv; l < LSEQ; l += 8) {
    const float* xr = x + (size_t)l * HDIM;
    float dot = 0.f, s = 0.f;
    for (int h = lane; h < HDIM; h += 32) { float v = xr[h]; dot += v * w[h]; s += v; }
    for (int o = 16; o; o >>= 1) { dot += __shfl_xor(dot, o); s += __shfl_xor(s, o); }
    if (lane == 0) attn[l] = (s != 0.f) ? dot : -1e9f;
  }
  __syncthreads();

  if (wv == 0) {
    float v0 = attn[lane], v1 = attn[lane + 32];
    float m = fmaxf(v0, v1);
    for (int o = 16; o; o >>= 1) m = fmaxf(m, __shfl_xor(m, o));
    float e0 = __expf(v0 - m), e1 = __expf(v1 - m);
    float s = e0 + e1;
    for (int o = 16; o; o >>= 1) s += __shfl_xor(s, o);
    attn[lane] = e0 / s; attn[lane + 32] = e1 / s;
    if (lane == 0) nrm = 0.f;
  }
  __syncthreads();

  float sq = 0.f;
  for (int h = tid; h < HDIM; h += 256) {
    float acc = 0.f;
    for (int l = 0; l < LSEQ; ++l) acc += attn[l] * x[(size_t)l * HDIM + h];
    pooled[h] = acc;
    sq += acc * acc;
  }
  atomicAdd(&nrm, sq);
  __syncthreads();
  if (tid == 0) pn[b] = sqrtf(nrm);
}

// per-item-row L2 norms, one wave per row
__global__ __launch_bounds__(256)
void inorm_kernel(const float* __restrict__ itA, const float* __restrict__ itB,
                  float* __restrict__ inA, float* __restrict__ inB) {
  const int wv = threadIdx.x >> 5, lane = threadIdx.x & 31;
  const int row = blockIdx.x * 8 + wv;
  const float* it = (blockIdx.y == 0 ? itA : itB) + (size_t)row * HDIM;
  float s = 0.f;
  for (int h = lane; h < HDIM; h += 32) { float v = it[h]; s += v * v; }
  for (int o = 16; o; o >>= 1) s += __shfl_xor(s, o);
  if (lane == 0) (blockIdx.y == 0 ? inA : inB)[row] = sqrtf(s);
}

// sim = pooled @ items^T : [256,2048] x [2048,50000]
// grid: x = M tiles (fastest) so both M-blocks of one item tile are adjacent
// in dispatch order -> the 820MB item stream goes through HBM exactly once,
// the second M-block hits L2.
__global__ __launch_bounds__(256)
void sim_kernel(const float* __restrict__ poolA, const float* __restrict__ poolB,
                const float* __restrict__ itA, const float* __restrict__ itB,
                float* __restrict__ sA, float* __restrict__ sB) {
  const float* P = blockIdx.z ? poolB : poolA;
  const float* I = blockIdx.z ? itB : itA;
  float* S = blockIdx.z ? sB : sA;
  gemm_nt_128x128<true>(P, I, S, NITEMS, HDIM, blockIdx.x, blockIdx.y);
}

// per-(row, domain) top-20 with cosine normalization; wave-0 parallel merge
__global__ __launch_bounds__(256)
void topk_kernel(const float* __restrict__ simA, const float* __restrict__ simB,
                 const float* __restrict__ pnA, const float* __restrict__ pnB,
                 const float* __restrict__ inA, const float* __restrict__ inB,
                 int* __restrict__ idxA, int* __restrict__ idxB) {
  const int b = blockIdx.x;
  const int dom = blockIdx.y;
  const float* sim = (dom == 0 ? simA : simB) + (size_t)b * NITEMS;
  const float* in  = dom == 0 ? inA : inB;
  const float rp   = 1.f / (dom == 0 ? pnA : pnB)[b];
  int* out = (dom == 0 ? idxA : idxB) + b * KTOP;

  __shared__ float cv[256 * KTOP];
  __shared__ int   ci[256 * KTOP];
  __shared__ int   heads[256];
  const int tid = threadIdx.x;

  float tv[KTOP]; int ti[KTOP];
#pragma unroll
  for (int j = 0; j < KTOP; ++j) { tv[j] = -1e30f; ti[j] = -1; }

  for (int i = tid; i < NITEMS; i += 256) {
    float v = sim[i] * rp / in[i];
    if (v > tv[KTOP - 1]) {
      int p = KTOP - 1;
      while (p > 0 && v > tv[p - 1]) { tv[p] = tv[p - 1]; ti[p] = ti[p - 1]; --p; }
      tv[p] = v; ti[p] = i;
    }
  }
#pragma unroll
  for (int j = 0; j < KTOP; ++j) { cv[tid * KTOP + j] = tv[j]; ci[tid * KTOP + j] = ti[j]; }
  heads[tid] = 0;
  __syncthreads();

  if (tid < 32) {
    const int lane = tid;
    const int offs = (dom == 0) ? 0 : NITEMS;   // +Anum for the B domain
    for (int r = 0; r < KTOP; ++r) {
      float bestv = -1e38f; int bestlist = 0x7fffffff; int besti = -1;
#pragma unroll
      for (int q = 0; q < 8; ++q) {
        int L = lane * 8 + q;
        int h = heads[L];
        float v = (h < KTOP) ? cv[L * KTOP + h] : -1e38f;
        if (v > bestv) { bestv = v; bestlist = L; besti = (h < KTOP) ? ci[L * KTOP + h] : -1; }
      }
      for (int o = 16; o; o >>= 1) {
        float ov = __shfl_xor(bestv, o);
        int   ol = __shfl_xor(bestlist, o);
        int   oi = __shfl_xor(besti, o);
        if (ov > bestv || (ov == bestv && ol < bestlist)) { bestv = ov; bestlist = ol; besti = oi; }
      }
      if (lane == (bestlist >> 3)) heads[bestlist]++;
      if (lane == 0) out[r] = besti + 1 + offs;
    }
  }
}

// domain select + gather data_embedding rows; indices (as float) head d_out
__global__ __launch_bounds__(256)
void gather_kernel(const int* __restrict__ idxA, const int* __restrict__ idxB,
                   const int* __restrict__ domains,
                   const float* __restrict__ data, float* __restrict__ out) {
  const int j = blockIdx.x, b = blockIdx.y;
  const int dom = domains[b];
  const int idx = (dom == 0 ? idxB : idxA)[b * KTOP + j];   // where(dom==0, B, A)
  const float4* src = (const float4*)(data + (size_t)idx * HDIM);
  float4* dst = (float4*)(out + (size_t)BSZ * KTOP + (size_t)(b * KTOP + j) * HDIM);
  for (int h = threadIdx.x; h < HDIM / 4; h += 256) dst[h] = src[h];
  if (threadIdx.x == 0) out[b * KTOP + j] = (float)idx;
}

extern "C" void kernel_launch(void* const* d_in, const int* in_sizes, int n_in,
                              void* d_out, int out_size, void* d_ws, size_t ws_size,
                              hipStream_t stream) {
  const float* gen   = (const float*)d_in[0];
  const float* projA = (const float*)d_in[1];
  const float* projB = (const float*)d_in[2];
  const float* itemA = (const float*)d_in[3];
  const float* itemB = (const float*)d_in[4];
  const float* data  = (const float*)d_in[5];
  const float* wA    = (const float*)d_in[6];
  const float* wB    = (const float*)d_in[7];
  const int*   doms  = (const int*)d_in[8];
  float* out = (float*)d_out;
  float* ws  = (float*)d_ws;

  const size_t GEN = (size_t)BSZ * LSEQ * HDIM;          // 33,554,432 floats
  float* Agen    = ws;
  float* Bgen    = ws + GEN;
  float* pooledA = ws + 2 * GEN;
  float* pooledB = pooledA + (size_t)BSZ * HDIM;
  float* pnA     = pooledB + (size_t)BSZ * HDIM;
  float* pnB     = pnA + BSZ;
  float* inA     = pnB + BSZ;
  float* inB     = inA + NITEMS;
  int*   idxA    = (int*)(inB + NITEMS);
  int*   idxB    = idxA + BSZ * KTOP;
  // sims alias the (dead after pooling) A_gen region: 2*12.8M < 33.5M floats
  float* simA    = ws;
  float* simB    = ws + (size_t)BSZ * NITEMS;

  dim3 blk(256);
  proj_kernel<<<dim3(HDIM / 128, (BSZ * LSEQ) / 128, 2), blk, 0, stream>>>(
      gen, projA, projB, Agen, Bgen);
  pool_kernel<<<dim3(BSZ, 2), blk, 0, stream>>>(
      Agen, Bgen, wA, wB, pooledA, pooledB, pnA, pnB);
  inorm_kernel<<<dim3(NITEMS / 8, 2), blk, 0, stream>>>(itemA, itemB, inA, inB);
  sim_kernel<<<dim3(BSZ / 128, (NITEMS + 127) / 128, 2), blk, 0, stream>>>(
      pooledA, pooledB, itemA, itemB, simA, simB);
  topk_kernel<<<dim3(BSZ, 2), blk, 0, stream>>>(
      simA, simB, pnA, pnB, inA, inB, idxA, idxB);
  gather_kernel<<<dim3(KTOP, BSZ), blk, 0, stream>>>(idxA, idxB, doms, data, out);
}
